// MahalaAttention_42537356099960
// MI455X (gfx1250) — compile-verified, matrix operand order checked by reading, not run on hardware
//
#include <hip/hip_runtime.h>
#include <hip/hip_bf16.h>

#define BATCH 4
#define HEADS 12
#define SEQ   512
#define HDIM  64
#define CDIM  768
#define C3    2304
#define BH    (BATCH*HEADS)   // 48
#define ROWS  (BATCH*SEQ)     // 2048

typedef __bf16 bf16;
typedef __attribute__((ext_vector_type(16))) __bf16 bf16x16;
typedef __attribute__((ext_vector_type(8)))  float  f32x8;

__device__ __forceinline__ bf16 f2bf(float f) {
  unsigned u = __float_as_uint(f);
  u += 0x7fffu + ((u >> 16) & 1u);          // round-to-nearest-even
  unsigned short h = (unsigned short)(u >> 16);
  return __builtin_bit_cast(bf16, h);
}
__device__ __forceinline__ float bf2f(bf16 b) {
  unsigned short h = __builtin_bit_cast(unsigned short, b);
  return __uint_as_float(((unsigned)h) << 16);
}
__device__ __forceinline__ bf16 bf_zero() {
  return __builtin_bit_cast(bf16, (unsigned short)0);
}

__device__ __forceinline__ f32x8 wmma_bf16(bf16x16 a, bf16x16 b, f32x8 c) {
  return __builtin_amdgcn_wmma_f32_16x16x32_bf16(false, a, false, b, (short)0, c,
                                                 false, false);
}

// 16x32 operand tile, element (r,k) at src[r*ld + k].
// Used as A (r = M-row) and as B when the B matrix is stored column(N)-major.
// ISA layout: lanes 0-15 -> r=lane, elems {K0..7, K16..23};
//             lanes 16-31 -> r=lane-16, elems {K8..15, K24..31}.
__device__ __forceinline__ bf16x16 load_frag_rmajor(const bf16* src, int ld) {
  int lane = threadIdx.x & 31;
  int r  = lane & 15;
  int k0 = (lane < 16) ? 0 : 8;
  const bf16* p = src + r * ld + k0;
  bf16x16 v;
#pragma unroll
  for (int i = 0; i < 8; ++i) v[i] = p[i];
#pragma unroll
  for (int i = 0; i < 8; ++i) v[8 + i] = p[16 + i];
  return v;
}

// B tile whose only nonzero column (n==0) is the vector e[j]; used to fold the
// softmax denominator Z = sum_j a[i,j]*E[j] into the WMMA as column 64.
__device__ __forceinline__ bf16x16 load_frag_ecol(const bf16* e, int j0) {
  int lane = threadIdx.x & 31;
  int k0 = (lane < 16) ? 0 : 8;
  bf16x16 v;
#pragma unroll
  for (int i = 0; i < 16; ++i) v[i] = bf_zero();
  if ((lane & 15) == 0) {
#pragma unroll
    for (int i = 0; i < 8; ++i) v[i] = e[j0 + k0 + i];
#pragma unroll
    for (int i = 0; i < 8; ++i) v[8 + i] = e[j0 + 16 + k0 + i];
  }
  return v;
}

// ---------------------------------------------------------------- converts
__global__ void k_cvt(const float* __restrict__ in, bf16* __restrict__ out, int n) {
  for (int i = blockIdx.x * blockDim.x + threadIdx.x; i < n;
       i += gridDim.x * blockDim.x)
    out[i] = f2bf(in[i]);
}

// ------------------------------------------------------------ QKV projection
// qkv[i,oc] = sum_d x[i,d]*qkv_w[oc,d]; scatter into q/k/v [B,H,N,hd] bf16,
// plus a transposed copy vT [B,H,hd,N] for the P@V stage.
__global__ __launch_bounds__(256) void k_qkv(const bf16* __restrict__ xb,
                                             const bf16* __restrict__ wb,
                                             bf16* __restrict__ qb,
                                             bf16* __restrict__ kb,
                                             bf16* __restrict__ vb,
                                             bf16* __restrict__ vtb) {
  int blk = blockIdx.x;
  int mt = blk & 127, nt = blk >> 7;          // 128 x 18
  int wid = threadIdx.x >> 5, lane = threadIdx.x & 31;
  int row0 = mt * 16;
  int col0 = nt * 128 + wid * 16;
  f32x8 acc = {};
  for (int k0 = 0; k0 < CDIM; k0 += 32) {
    bf16x16 a = load_frag_rmajor(xb + (size_t)row0 * CDIM + k0, CDIM);
    bf16x16 b = load_frag_rmajor(wb + (size_t)col0 * CDIM + k0, CDIM);
    acc = wmma_bf16(a, b, acc);
  }
  int n = lane & 15, m0 = (lane < 16) ? 0 : 8;
  int oc = col0 + n;
  int part = oc / CDIM;                       // 0=q 1=k 2=v (uniform per tile)
  int rem = oc - part * CDIM;
  int h = rem >> 6, dd = rem & 63;
  bf16* dst = (part == 0) ? qb : (part == 1 ? kb : vb);
#pragma unroll
  for (int r = 0; r < 8; ++r) {
    int i = row0 + m0 + r;
    int bi = i >> 9, nn = i & 511;
    bf16 val = f2bf(acc[r]);
    dst[(((size_t)bi * HEADS + h) * SEQ + nn) * HDIM + dd] = val;
    if (part == 2)
      vtb[(((size_t)bi * HEADS + h) * HDIM + dd) * SEQ + nn] = val;
  }
}

// ----------------------------------------------- base scores -> e = exp(s-max)
__global__ __launch_bounds__(256) void k_escore(const bf16* __restrict__ qb,
                                                const bf16* __restrict__ kb,
                                                bf16* __restrict__ eb) {
  __shared__ float s_s[16 * 520];
  __shared__ float s_red[256];
  __shared__ float s_max[16];
  int blk = blockIdx.x;
  int bh = blk >> 5;
  int i0 = (blk & 31) * 16;
  const bf16* q  = qb + (size_t)bh * SEQ * HDIM;
  const bf16* kk = kb + (size_t)bh * SEQ * HDIM;
  int wid = threadIdx.x >> 5, lane = threadIdx.x & 31;

  f32x8 acc[4] = {};
  for (int d0 = 0; d0 < HDIM; d0 += 32) {
    bf16x16 a = load_frag_rmajor(q + (size_t)i0 * HDIM + d0, HDIM);
#pragma unroll
    for (int c = 0; c < 4; ++c) {
      int jt = wid * 4 + c;
      bf16x16 b = load_frag_rmajor(kk + (size_t)(jt * 16) * HDIM + d0, HDIM);
      acc[c] = wmma_bf16(a, b, acc[c]);
    }
  }
  int n = lane & 15, m0 = (lane < 16) ? 0 : 8;
#pragma unroll
  for (int c = 0; c < 4; ++c)
#pragma unroll
    for (int r = 0; r < 8; ++r)
      s_s[(m0 + r) * 520 + (wid * 4 + c) * 16 + n] = acc[c][r] * 0.125f;
  __syncthreads();
  {   // two-stage parallel rowmax: 16 threads/row, 32 cols each
    int r = threadIdx.x >> 4, c0 = (threadIdx.x & 15) * 32;
    float m = -1e30f;
    for (int c = 0; c < 32; ++c) m = fmaxf(m, s_s[r * 520 + c0 + c]);
    s_red[threadIdx.x] = m;
  }
  __syncthreads();
  if (threadIdx.x < 16) {
    float m = -1e30f;
    for (int t = 0; t < 16; ++t) m = fmaxf(m, s_red[threadIdx.x * 16 + t]);
    s_max[threadIdx.x] = m;
  }
  __syncthreads();
  bf16* erow = eb + (size_t)bh * SEQ * SEQ;
  for (int idx = threadIdx.x; idx < 16 * SEQ; idx += 256) {
    int r = idx >> 9, cc = idx & 511;
    float e = __expf(s_s[r * 520 + cc] - s_max[r]);
    erow[(size_t)(i0 + r) * SEQ + cc] = f2bf(e);
  }
}

// ------------------------------------------------------- perturbation sweep
// One block per (b,h,ei). Hp = (e @ Ep*V)/(e @ Ep), Hm analog; accumulate
// sum|Hp-Hm| / (2 t N) into W[b,h,ei]. Z folded into WMMA as column tile 4.
__global__ __launch_bounds__(256) void k_perturb(const bf16* __restrict__ kb,
                                                 const bf16* __restrict__ vb,
                                                 const bf16* __restrict__ eb,
                                                 float* __restrict__ Wbuf) {
  __shared__ bf16  s_mtp[HDIM * SEQ];   // (Ep*V)^T  64KB
  __shared__ bf16  s_mtm[HDIM * SEQ];   // (Em*V)^T  64KB
  __shared__ float s_ep[SEQ];
  __shared__ float s_em[SEQ];
  __shared__ bf16  s_epb[SEQ];
  __shared__ bf16  s_emb[SEQ];
  __shared__ float s_zp[8 * 16];
  __shared__ float s_zm[8 * 16];
  int blk = blockIdx.x;
  int ei = blk & 63, bh = blk >> 6;
  const bf16* kk = kb + (size_t)bh * SEQ * HDIM;
  const bf16* vv = vb + (size_t)bh * SEQ * HDIM;
  const bf16* ee = eb + (size_t)bh * SEQ * SEQ;

  for (int j = threadIdx.x; j < SEQ; j += 256) {
    float d = 0.0078125f * bf2f(kk[j * HDIM + ei]);  // t*scale*k[j,ei]
    float ep = __expf(d), em = __expf(-d);
    s_ep[j] = ep;  s_em[j] = em;
    s_epb[j] = f2bf(ep);  s_emb[j] = f2bf(em);
  }
  __syncthreads();
  for (int idx = threadIdx.x; idx < SEQ * HDIM; idx += 256) {
    int c = idx & 63, j = idx >> 6;
    float v = bf2f(vv[idx]);                          // vv[j*64+c]
    s_mtp[c * SEQ + j] = f2bf(v * s_ep[j]);
    s_mtm[c * SEQ + j] = f2bf(v * s_em[j]);
  }
  __syncthreads();

  int wid = threadIdx.x >> 5, lane = threadIdx.x & 31;
  float wsum = 0.f;
  for (int it = 0; it < 4; ++it) {
    int i0 = wid * 16 + it * 128;
    f32x8 ap[5] = {}, am[5] = {};
    for (int j0 = 0; j0 < SEQ; j0 += 32) {
      bf16x16 a = load_frag_rmajor(ee + (size_t)i0 * SEQ + j0, SEQ);
#pragma unroll
      for (int c = 0; c < 4; ++c) {
        bf16x16 bp = load_frag_rmajor(s_mtp + (c * 16) * SEQ + j0, SEQ);
        bf16x16 bm = load_frag_rmajor(s_mtm + (c * 16) * SEQ + j0, SEQ);
        ap[c] = wmma_bf16(a, bp, ap[c]);
        am[c] = wmma_bf16(a, bm, am[c]);
      }
      bf16x16 bep = load_frag_ecol(s_epb, j0);
      bf16x16 bem = load_frag_ecol(s_emb, j0);
      ap[4] = wmma_bf16(a, bep, ap[4]);
      am[4] = wmma_bf16(a, bem, am[4]);
    }
    int m0 = (lane < 16) ? 0 : 8;
    if ((lane & 15) == 0) {      // col 64 of tile 4 = Z values
#pragma unroll
      for (int r = 0; r < 8; ++r) {
        s_zp[wid * 16 + m0 + r] = ap[4][r];
        s_zm[wid * 16 + m0 + r] = am[4][r];
      }
    }
#pragma unroll
    for (int r = 0; r < 8; ++r) {
      float izp = 1.f / s_zp[wid * 16 + m0 + r];
      float izm = 1.f / s_zm[wid * 16 + m0 + r];
#pragma unroll
      for (int c = 0; c < 4; ++c)
        wsum += fabsf(ap[c][r] * izp - am[c][r] * izm);
    }
  }
#pragma unroll
  for (int off = 16; off > 0; off >>= 1) wsum += __shfl_xor(wsum, off, 32);
  if (lane == 0) atomicAdd(&Wbuf[bh * 64 + ei], wsum * 0.015625f); // /(2 t N)
}

// ----------------------------------------------- final weighted attention
__global__ __launch_bounds__(256) void k_attn(const bf16* __restrict__ qb,
                                              const bf16* __restrict__ kb,
                                              const bf16* __restrict__ vtb,
                                              const float* __restrict__ Wbuf,
                                              bf16* __restrict__ ob) {
  __shared__ float s_s[16 * 520];
  __shared__ bf16  s_p[16 * 512];
  __shared__ float s_w[HDIM];
  __shared__ float s_red[256];
  __shared__ float s_max[16];
  __shared__ float s_sum[16];
  __shared__ float s_o[16 * HDIM];
  int blk = blockIdx.x;
  int bh = blk >> 5;
  int i0 = (blk & 31) * 16;
  int bi = bh / HEADS, h = bh - bi * HEADS;
  const bf16* q  = qb + (size_t)bh * SEQ * HDIM;
  const bf16* kk = kb + (size_t)bh * SEQ * HDIM;
  const bf16* vt = vtb + (size_t)bh * HDIM * SEQ;
  if (threadIdx.x < HDIM) s_w[threadIdx.x] = Wbuf[bh * HDIM + threadIdx.x];
  __syncthreads();

  int wid = threadIdx.x >> 5, lane = threadIdx.x & 31;
  int rowq = lane & 15, k0l = (lane < 16) ? 0 : 8;
  f32x8 acc[4] = {};
  for (int d0 = 0; d0 < HDIM; d0 += 32) {
    bf16x16 a;                                   // A frag of q*diag(W)
    const bf16* qr = q + (size_t)(i0 + rowq) * HDIM + d0;
#pragma unroll
    for (int i2 = 0; i2 < 8; ++i2) {
      int d = k0l + i2;
      a[i2] = f2bf(bf2f(qr[d]) * s_w[d0 + d]);
    }
#pragma unroll
    for (int i2 = 0; i2 < 8; ++i2) {
      int d = 16 + k0l + i2;
      a[8 + i2] = f2bf(bf2f(qr[d]) * s_w[d0 + d]);
    }
#pragma unroll
    for (int c = 0; c < 4; ++c) {
      int jt = wid * 4 + c;
      bf16x16 b = load_frag_rmajor(kk + (size_t)(jt * 16) * HDIM + d0, HDIM);
      acc[c] = wmma_bf16(a, b, acc[c]);
    }
  }
  int n = lane & 15, m0 = (lane < 16) ? 0 : 8;
#pragma unroll
  for (int c = 0; c < 4; ++c)
#pragma unroll
    for (int r = 0; r < 8; ++r)
      s_s[(m0 + r) * 520 + (wid * 4 + c) * 16 + n] = acc[c][r] * 0.125f;
  __syncthreads();
  {   // parallel rowmax
    int r = threadIdx.x >> 4, c0 = (threadIdx.x & 15) * 32;
    float m = -1e30f;
    for (int c = 0; c < 32; ++c) m = fmaxf(m, s_s[r * 520 + c0 + c]);
    s_red[threadIdx.x] = m;
  }
  __syncthreads();
  if (threadIdx.x < 16) {
    float m = -1e30f;
    for (int t = 0; t < 16; ++t) m = fmaxf(m, s_red[threadIdx.x * 16 + t]);
    s_max[threadIdx.x] = m;
    s_sum[threadIdx.x] = 0.f;
  }
  __syncthreads();
  {   // exp + rowsum
    int r = threadIdx.x >> 4;
    int c0 = (threadIdx.x & 15) * 32;
    float part = 0.f;
    for (int c = 0; c < 32; ++c) {
      float e = __expf(s_s[r * 520 + c0 + c] - s_max[r]);
      s_p[r * 512 + c0 + c] = f2bf(e);
      part += e;
    }
    atomicAdd(&s_sum[r], part);
  }
  for (int idx = threadIdx.x; idx < 16 * HDIM; idx += 256) s_o[idx] = 0.f;
  __syncthreads();
  // P @ V : K split across 8 waves (64 keys each), LDS float reduction.
  // B operand from transposed V copy -> contiguous 16B loads.
  f32x8 pv[4] = {};
  int j0w = wid * 64;
  for (int ks = 0; ks < 2; ++ks) {
    int j0 = j0w + ks * 32;
    bf16x16 a = load_frag_rmajor((const bf16*)s_p + j0, 512);
#pragma unroll
    for (int c = 0; c < 4; ++c) {
      bf16x16 b = load_frag_rmajor(vt + (size_t)(c * 16) * SEQ + j0, SEQ);
      pv[c] = wmma_bf16(a, b, pv[c]);
    }
  }
#pragma unroll
  for (int c = 0; c < 4; ++c)
#pragma unroll
    for (int r = 0; r < 8; ++r)
      atomicAdd(&s_o[(m0 + r) * HDIM + c * 16 + n], pv[c][r]);
  __syncthreads();
  for (int idx = threadIdx.x; idx < 16 * HDIM; idx += 256) {
    int m = idx >> 6, cc = idx & 63;
    float val = s_o[idx] / s_sum[m];
    ob[((size_t)bi * SEQ + i0 + m) * CDIM + h * HDIM + cc] = f2bf(val);
  }
}

// ------------------------------------------------------- output projection
__global__ __launch_bounds__(256) void k_proj(const bf16* __restrict__ ab,
                                              const bf16* __restrict__ wb,
                                              const float* __restrict__ bias,
                                              float* __restrict__ out) {
  int blk = blockIdx.x;
  int mt = blk & 127, nt = blk >> 7;          // 128 x 6
  int wid = threadIdx.x >> 5, lane = threadIdx.x & 31;
  int row0 = mt * 16, col0 = nt * 128 + wid * 16;
  f32x8 acc = {};
  for (int k0 = 0; k0 < CDIM; k0 += 32) {
    bf16x16 a = load_frag_rmajor(ab + (size_t)row0 * CDIM + k0, CDIM);
    bf16x16 b = load_frag_rmajor(wb + (size_t)col0 * CDIM + k0, CDIM);
    acc = wmma_bf16(a, b, acc);
  }
  int n = lane & 15, m0 = (lane < 16) ? 0 : 8;
  int oc = col0 + n;
  float bv = bias[oc];
#pragma unroll
  for (int r = 0; r < 8; ++r)
    out[(size_t)(row0 + m0 + r) * CDIM + oc] = acc[r] + bv;
}

extern "C" void kernel_launch(void* const* d_in, const int* in_sizes, int n_in,
                              void* d_out, int out_size, void* d_ws, size_t ws_size,
                              hipStream_t stream) {
  (void)in_sizes; (void)n_in; (void)out_size; (void)ws_size;
  const float* x     = (const float*)d_in[0];
  const float* qkvw  = (const float*)d_in[1];
  const float* projw = (const float*)d_in[2];
  const float* projb = (const float*)d_in[3];
  float* out = (float*)d_out;

  char* ws = (char*)d_ws;
  size_t o = 0;
  auto take = [&](size_t bytes) -> char* {
    char* p = ws + o;
    o += (bytes + 255) & ~(size_t)255;
    return p;
  };
  bf16* xb   = (bf16*)take((size_t)ROWS * CDIM * 2);
  bf16* wqb  = (bf16*)take((size_t)C3 * CDIM * 2);
  bf16* wpb  = (bf16*)take((size_t)CDIM * CDIM * 2);
  bf16* qb   = (bf16*)take((size_t)BH * SEQ * HDIM * 2);
  bf16* kb   = (bf16*)take((size_t)BH * SEQ * HDIM * 2);
  bf16* vb   = (bf16*)take((size_t)BH * SEQ * HDIM * 2);
  bf16* vtb  = (bf16*)take((size_t)BH * HDIM * SEQ * 2);
  bf16* eb   = (bf16*)take((size_t)BH * SEQ * SEQ * 2);
  bf16* ab   = (bf16*)take((size_t)ROWS * CDIM * 2);
  float* Wbuf = (float*)take((size_t)BH * HDIM * 4);

  hipMemsetAsync(Wbuf, 0, (size_t)BH * HDIM * 4, stream);
  k_cvt<<<2048, 256, 0, stream>>>(x, xb, ROWS * CDIM);
  k_cvt<<<2048, 256, 0, stream>>>(qkvw, wqb, C3 * CDIM);
  k_cvt<<<1024, 256, 0, stream>>>(projw, wpb, CDIM * CDIM);
  k_qkv<<<2304, 256, 0, stream>>>(xb, wqb, qb, kb, vb, vtb);
  k_escore<<<BH * 32, 256, 0, stream>>>(qb, kb, eb);
  k_perturb<<<BH * HDIM, 256, 0, stream>>>(kb, vb, eb, Wbuf);
  k_attn<<<BH * 32, 256, 0, stream>>>(qb, kb, vtb, Wbuf, ab);
  k_proj<<<768, 256, 0, stream>>>(ab, wpb, projb, out);
}
